// VisionTransformerMoE_72765335929166
// MI455X (gfx1250) — compile-verified, matrix operand order checked by reading, not run on hardware
//
#include <hip/hip_runtime.h>

// ---------------- model dims ----------------
#define LL    4
#define DD    384
#define HID   1536
#define NHD   6
#define EE    8
#define KK    2
#define IMGS  224
#define NCLS  1000
#define BB    8
#define SS    197              // tokens per image (cls + 14*14)
#define TT    (BB*SS)          // 1576
#define PD    768              // 3*16*16 patch feature dim
#define NP    196              // patches per image
#define QKVD  1152
#define HDIM  64               // head dim

typedef __attribute__((ext_vector_type(16))) __bf16 v16bf;
typedef __attribute__((ext_vector_type(8)))  float  v8f;

union Frag { unsigned int ui[8]; v16bf bf; };

// Pack two f32 -> two bf16 in one dword with a single v_perm_b32.
// Round-to-nearest (ties away) via +0x8000, then take bytes [3:2] of each.
__device__ inline unsigned int pack2bf(float a, float b) {
  unsigned int ua = __builtin_bit_cast(unsigned int, a) + 0x8000u;
  unsigned int ub = __builtin_bit_cast(unsigned int, b) + 0x8000u;
  // v_perm_b32: bytes 0-3 from S1(=ua), 4-7 from S0(=ub);
  // sel 0x07060302 -> {ub.b3, ub.b2, ua.b3, ua.b2} = (bf16(b)<<16)|bf16(a)
  return __builtin_amdgcn_perm(ub, ua, 0x07060302u);
}
__device__ inline float gelu_f(float x) {       // exact gelu (approximate=False)
  return 0.5f * x * (1.0f + erff(x * 0.70710678118654752f));
}

// ---------------- WMMA GEMM ----------------
// C[M,N] = A[M,K] @ W[K,N] (+bias, epilogue per MODE), bf16 WMMA, fp32 accum.
// 64x64x32 tile, 128 threads = 4 waves, wave computes 2x2 of 16x16 WMMA tiles.
// Software-pipelined: next K-slab global->regs overlaps WMMAs of current slab.
#define BM 64
#define BN 64
#define BK 32
#define LDK (BK + 4)           // 36 shorts = 72B row stride (8B-aligned, odd bank stride)

enum { GM_BIAS = 0, GM_GATHER_GELU = 2, GM_SCATTER_MOE = 3 };

template <int MODE>
__global__ __launch_bounds__(128) void vit_gemm(
    const float* __restrict__ A, const float* __restrict__ W,
    const float* __restrict__ bias, float* __restrict__ C,
    int M, int N, int Kd, int lda, int ldc,
    const int* __restrict__ tok, const float* __restrict__ gate,
    const int* __restrict__ cnt, const float* __restrict__ bias0,
    const float* __restrict__ b2, float* __restrict__ moe_out)
{
  int mlim = M;
  if (MODE == GM_GATHER_GELU || MODE == GM_SCATTER_MOE) {
    int c0 = *cnt; mlim = (c0 < M) ? c0 : M;     // routed-token count for this expert
  }
  const int tm0 = blockIdx.x * BM;
  if (tm0 >= mlim) return;                        // uniform early-exit, before barriers
  const int tn0 = blockIdx.y * BN;

  __shared__ unsigned short As[BM][LDK];          // [row][k]
  __shared__ unsigned short Bs[BN][LDK];          // [col][k] (K-contiguous per column)

  const int tid  = threadIdx.x;
  const int lane = tid & 31;
  const int wave = tid >> 5;
  const int wm   = (wave >> 1) * 32;
  const int wn   = (wave & 1) * 32;
  const int l16  = lane & 15;
  const int hi8  = (lane & 16) >> 1;              // 0 or 8 (K-half select per ISA layout)

  v8f acc[2][2] = {};

  // A-tile loader: thread covers half a row (16 contiguous floats)
  const int arow = tid >> 1;                      // 0..63
  const int aks  = (tid & 1) << 4;                // 0 or 16
  const int gr   = tm0 + arow;
  const int grc  = (gr < mlim) ? gr : (mlim - 1); // clamped -> always-valid row
  const float amask = (gr < mlim) ? 1.0f : 0.0f;  // zero out padded rows (branch-free)
  const long Abase =
      ((MODE == GM_GATHER_GELU) ? (long)tok[grc] : (long)grc) * (long)lda + aks;

  // B-tile loader: thread covers one column, 16 K-values (coalesced across threads)
  const int bcol = tid >> 1;
  const int bks  = (tid & 1) << 4;
  const float* __restrict__ Wp = W + (long)bks * N + tn0 + bcol;

  float areg[16], breg[16];
  auto load_regs = [&](int k0) {
    const float4* __restrict__ Ap = (const float4*)(A + Abase + k0);
#pragma unroll
    for (int i = 0; i < 4; ++i) {
      float4 q = Ap[i];
      areg[4*i+0] = q.x; areg[4*i+1] = q.y; areg[4*i+2] = q.z; areg[4*i+3] = q.w;
    }
#pragma unroll
    for (int j = 0; j < 16; ++j) breg[j] = Wp[(long)(k0 + j) * N];
  };

  load_regs(0);
  for (int k0 = 0; k0 < Kd; k0 += BK) {
    // regs -> LDS (packed bf16 pairs via v_perm_b32, b32/b64 stores)
    unsigned int* __restrict__ ar = (unsigned int*)&As[arow][aks];
    unsigned int* __restrict__ br = (unsigned int*)&Bs[bcol][bks];
#pragma unroll
    for (int p = 0; p < 8; ++p) {
      ar[p] = pack2bf(areg[2*p] * amask, areg[2*p+1] * amask);
      br[p] = pack2bf(breg[2*p],         breg[2*p+1]);
    }
    __syncthreads();

    if (k0 + BK < Kd) {                           // prefetch next slab into regs;
      load_regs(k0 + BK);                         // its wait overlaps the WMMAs below
      __builtin_prefetch(&Wp[(long)(k0 + 2*BK) * N], 0, 1);
    }

    // Fragment loads per CDNA5 16-bit A-matrix 16x32 layout:
    // VGPR p<4 -> K = 2p (+8 for lanes 16..31); p>=4 -> K = 16+2(p-4) (+8)
    Frag a0, a1, b0, b1;
#pragma unroll
    for (int p = 0; p < 8; ++p) {
      const int kb = ((p < 4) ? (p << 1) : (16 + ((p - 4) << 1))) + hi8;
      a0.ui[p] = *(const unsigned int*)&As[wm      + l16][kb];
      a1.ui[p] = *(const unsigned int*)&As[wm + 16 + l16][kb];
      b0.ui[p] = *(const unsigned int*)&Bs[wn      + l16][kb];
      b1.ui[p] = *(const unsigned int*)&Bs[wn + 16 + l16][kb];
    }
    acc[0][0] = __builtin_amdgcn_wmma_f32_16x16x32_bf16(false, a0.bf, false, b0.bf, (short)0, acc[0][0], false, false);
    acc[0][1] = __builtin_amdgcn_wmma_f32_16x16x32_bf16(false, a0.bf, false, b1.bf, (short)0, acc[0][1], false, false);
    acc[1][0] = __builtin_amdgcn_wmma_f32_16x16x32_bf16(false, a1.bf, false, b0.bf, (short)0, acc[1][0], false, false);
    acc[1][1] = __builtin_amdgcn_wmma_f32_16x16x32_bf16(false, a1.bf, false, b1.bf, (short)0, acc[1][1], false, false);
    __syncthreads();
  }

  // C/D 16x16 f32 layout: lanes 0-15: N=lane, M=r; lanes 16-31: N=lane-16, M=8+r
#pragma unroll
  for (int tm = 0; tm < 2; ++tm) {
#pragma unroll
    for (int tn = 0; tn < 2; ++tn) {
#pragma unroll
      for (int r = 0; r < 8; ++r) {
        const int row = tm0 + wm + tm * 16 + r + hi8;
        const int col = tn0 + wn + tn * 16 + l16;
        if (row >= mlim) continue;
        float v = acc[tm][tn][r];
        if (MODE == GM_BIAS) {
          C[(long)row * ldc + col] = v + bias[col];
        } else if (MODE == GM_GATHER_GELU) {
          C[(long)row * ldc + col] = gelu_f(v + bias[col]);
        } else {  // GM_SCATTER_MOE: out[tok] += gate*(y + b2 - bias0)
          const int   tt = tok[row];
          const float g  = gate[row];
          atomicAdd(&moe_out[(long)tt * ldc + col], g * (v + b2[col] - bias0[col]));
        }
      }
    }
  }
}

// ---------------- small kernels ----------------
__global__ void vit_patchify(const float* __restrict__ x, float* __restrict__ patches) {
  int i = blockIdx.x * blockDim.x + threadIdx.x;
  if (i >= BB * NP * PD) return;
  int f = i % PD, bp = i / PD;
  int b = bp / NP, pch = bp % NP;
  int hy = pch / 14, hx = pch % 14;
  int c = f / 256, r = f % 256;
  int py = r / 16, px = r % 16;
  patches[i] = x[(((long)b * 3 + c) * IMGS + hy * 16 + py) * IMGS + hx * 16 + px];
}

__global__ void vit_assemble(const float* __restrict__ Yp, const float* __restrict__ cls,
                             const float* __restrict__ pos, float* __restrict__ t) {
  int i = blockIdx.x * blockDim.x + threadIdx.x;
  if (i >= TT * DD) return;
  int d = i % DD, bs = i / DD;
  int b = bs / SS, s = bs % SS;
  float v = (s == 0) ? cls[d] : Yp[((long)b * NP + (s - 1)) * DD + d];
  t[i] = v + pos[s * DD + d];
}

// fused scores + softmax for one (b,h,q) row
__global__ void vit_attn_scores(const float* __restrict__ qkv, float* __restrict__ att) {
  const int bid = blockIdx.x;
  const int q = bid % SS;
  const int h = (bid / SS) % NHD;
  const int b = bid / (SS * NHD);
  const int tid = threadIdx.x;
  __shared__ float red[256];
  const float* qrow = qkv + ((long)(b * SS + q)) * QKVD + h * HDIM;
  float val = -1e30f;
  if (tid < SS) {
    const float* krow = qkv + ((long)(b * SS + tid)) * QKVD + DD + h * HDIM;
    float s = 0.0f;
#pragma unroll 8
    for (int d = 0; d < HDIM; ++d) s += qrow[d] * krow[d];
    val = s * 0.125f;                               // 1/sqrt(64)
  }
  red[tid] = val; __syncthreads();
  for (int s = 128; s > 0; s >>= 1) { if (tid < s) red[tid] = fmaxf(red[tid], red[tid + s]); __syncthreads(); }
  const float mx = red[0]; __syncthreads();
  float e = (tid < SS) ? __expf(val - mx) : 0.0f;
  red[tid] = e; __syncthreads();
  for (int s = 128; s > 0; s >>= 1) { if (tid < s) red[tid] += red[tid + s]; __syncthreads(); }
  if (tid < SS) att[((long)(b * NHD + h) * SS + q) * SS + tid] = e / red[0];
}

__global__ void vit_av(const float* __restrict__ att, const float* __restrict__ qkv,
                       float* __restrict__ o) {
  const int b = blockIdx.x / SS, q = blockIdx.x % SS;
  const int d = threadIdx.x;          // 0..383
  const int h = d >> 6;
  const float* arow = att + ((long)(b * NHD + h) * SS + q) * SS;
  float s = 0.0f;
  for (int k = 0; k < SS; ++k)
    s += arow[k] * qkv[((long)(b * SS + k)) * QKVD + 2 * DD + h * HDIM + (d & 63)];
  o[((long)(b * SS + q)) * DD + d] = s;
}

// out = LN(x (+ add)) * g + b ; one wave32 per row, 12 elems/lane
__global__ void vit_add_ln(const float* __restrict__ xin, const float* __restrict__ add,
                           const float* __restrict__ g, const float* __restrict__ bta,
                           float* __restrict__ out, int rows) {
  const int row = blockIdx.x * 8 + (threadIdx.x >> 5);
  const int lane = threadIdx.x & 31;
  if (row >= rows) return;
  const float* xr = xin + (long)row * DD;
  const float* ar = add ? add + (long)row * DD : nullptr;
  float v[12];
  float s = 0.0f;
#pragma unroll
  for (int i = 0; i < 12; ++i) {
    float t = xr[lane + i * 32];
    if (ar) t += ar[lane + i * 32];
    v[i] = t; s += t;
  }
  for (int o2 = 16; o2; o2 >>= 1) s += __shfl_xor(s, o2, 32);
  const float m = s / (float)DD;
  float var = 0.0f;
#pragma unroll
  for (int i = 0; i < 12; ++i) { float d = v[i] - m; var += d * d; }
  for (int o2 = 16; o2; o2 >>= 1) var += __shfl_xor(var, o2, 32);
  const float rstd = rsqrtf(var / (float)DD + 1e-5f);
#pragma unroll
  for (int i = 0; i < 12; ++i)
    out[(long)row * DD + lane + i * 32] = (v[i] - m) * rstd * g[lane + i * 32] + bta[lane + i * 32];
}

// router: logits, stable top-2, softmax gates, per-expert token lists, occupancy
__global__ void vit_router(const float* __restrict__ t, const float* __restrict__ rw,
                           const float* __restrict__ rb, int* __restrict__ cnt,
                           int* __restrict__ tok_list, float* __restrict__ gate_list,
                           int* __restrict__ occ, float* __restrict__ gates2,
                           int* __restrict__ idx2) {
  const int tk = blockIdx.x;
  const int lane = threadIdx.x;
  float logit = -1e30f;
  if (lane < EE) {
    const float* xr = t + (long)tk * DD;
    float s = rb[lane];
    for (int d = 0; d < DD; ++d) s += xr[d] * rw[d * EE + lane];
    logit = s;   // NOTE: jax PRNG gating noise (0.01 scale) not reproducible on device
  }
  float lv[EE];
#pragma unroll
  for (int e = 0; e < EE; ++e) lv[e] = __shfl(logit, e, 32);
  if (lane == 0) {
    float best = -1e30f, sec = -1e30f; int bi = 0, si = 0;
#pragma unroll
    for (int e = 0; e < EE; ++e) {
      if (lv[e] > best) { sec = best; si = bi; best = lv[e]; bi = e; }
      else if (lv[e] > sec) { sec = lv[e]; si = e; }
    }
    const float e1 = __expf(sec - best);
    const float g0 = 1.0f / (1.0f + e1);
    const float g1 = e1 / (1.0f + e1);
    int p0 = atomicAdd(&cnt[bi], 1);
    tok_list[bi * TT + p0] = tk; gate_list[bi * TT + p0] = g0;
    int p1 = atomicAdd(&cnt[si], 1);
    tok_list[si * TT + p1] = tk; gate_list[si * TT + p1] = g1;
    atomicOr(&occ[bi], 1); atomicOr(&occ[EE + si], 1);
    gates2[tk * 2] = g0; gates2[tk * 2 + 1] = g1;
    idx2[tk * 2] = bi;  idx2[tk * 2 + 1] = si;
  }
}

// bias0[e,d] = gelu(b1[e]) @ w2[e] + b2[e]
__global__ void vit_bias0(const float* __restrict__ b1, const float* __restrict__ w2,
                          const float* __restrict__ b2, float* __restrict__ bias0) {
  int i = blockIdx.x * blockDim.x + threadIdx.x;
  if (i >= EE * DD) return;
  int e = i / DD, d = i % DD;
  float s = b2[e * DD + d];
  const float* b1e = b1 + e * HID;
  const float* w2e = w2 + (long)e * HID * DD;
  for (int h = 0; h < HID; ++h) s += gelu_f(b1e[h]) * w2e[(long)h * DD + d];
  bias0[i] = s;
}

__global__ void vit_biassum(const int* __restrict__ occ, const float* __restrict__ bias0,
                            float* __restrict__ bias_sum) {
  int i = blockIdx.x * blockDim.x + threadIdx.x;
  if (i >= KK * DD) return;
  int k = i / DD, d = i % DD;
  float s = 0.0f;
  for (int e = 0; e < EE; ++e) if (occ[k * EE + e]) s += bias0[e * DD + d];
  bias_sum[i] = s;
}

__global__ void vit_moe_biasadd(const float* __restrict__ gates2, const float* __restrict__ bias_sum,
                                float* __restrict__ moe_out) {
  int i = blockIdx.x * blockDim.x + threadIdx.x;
  if (i >= TT * DD) return;
  int t = i / DD, d = i % DD;
  moe_out[i] += gates2[t * 2] * bias_sum[d] + gates2[t * 2 + 1] * bias_sum[DD + d];
}

__global__ void vit_head(const float* __restrict__ t, const float* __restrict__ fw,
                         const float* __restrict__ fb, float* __restrict__ out) {
  int i = blockIdx.x * blockDim.x + threadIdx.x;
  if (i >= BB * NCLS) return;
  int b = i / NCLS, c = i % NCLS;
  const float* xr = t + (long)b * SS * DD;     // cls row
  float s = fb[c];
  for (int d = 0; d < DD; ++d) s += xr[d] * fw[(long)d * NCLS + c];
  out[i] = s;
}

// ---------------- host orchestration ----------------
extern "C" void kernel_launch(void* const* d_in, const int* in_sizes, int n_in,
                              void* d_out, int out_size, void* d_ws, size_t ws_size,
                              hipStream_t stream) {
  const float* x        = (const float*)d_in[0];
  const float* patch_w  = (const float*)d_in[1];
  const float* patch_b  = (const float*)d_in[2];
  const float* cls_tok  = (const float*)d_in[3];
  const float* pos      = (const float*)d_in[4];
  const float* qkv_w    = (const float*)d_in[5];
  const float* qkv_b    = (const float*)d_in[6];
  const float* proj_w   = (const float*)d_in[7];
  const float* proj_b   = (const float*)d_in[8];
  const float* ln1_g    = (const float*)d_in[9];
  const float* ln1_b    = (const float*)d_in[10];
  const float* ln2_g    = (const float*)d_in[11];
  const float* ln2_b    = (const float*)d_in[12];
  const float* router_w = (const float*)d_in[13];
  const float* router_b = (const float*)d_in[14];
  const float* w1       = (const float*)d_in[15];
  const float* b1       = (const float*)d_in[16];
  const float* w2       = (const float*)d_in[17];
  const float* b2       = (const float*)d_in[18];
  const float* norm_g   = (const float*)d_in[19];
  const float* norm_b   = (const float*)d_in[20];
  const float* fc_w     = (const float*)d_in[21];
  const float* fc_b     = (const float*)d_in[22];
  float* out = (float*)d_out;
  (void)in_sizes; (void)n_in; (void)out_size; (void)ws_size;

  char* wp = (char*)d_ws;
  auto alloc = [&](size_t bytes) -> char* {
    char* r = wp; wp += (bytes + 255) & ~(size_t)255; return r;
  };
  float* patches   = (float*)alloc((size_t)BB * NP * PD * 4);
  float* Yp        = (float*)alloc((size_t)BB * NP * DD * 4);
  float* t         = (float*)alloc((size_t)TT * DD * 4);
  float* qkv       = (float*)alloc((size_t)TT * QKVD * 4);
  float* att       = (float*)alloc((size_t)BB * NHD * SS * SS * 4);
  float* obuf      = (float*)alloc((size_t)TT * DD * 4);
  float* abuf      = (float*)alloc((size_t)TT * DD * 4);
  float* moe_out   = (float*)alloc((size_t)TT * DD * 4);
  float* h_buf     = (float*)alloc((size_t)TT * HID * 4);
  int*   cnt       = (int*)alloc(EE * 4);
  int*   occ       = (int*)alloc(2 * EE * 4);
  int*   tok_list  = (int*)alloc((size_t)EE * TT * 4);
  float* gate_list = (float*)alloc((size_t)EE * TT * 4);
  float* gates2    = (float*)alloc((size_t)TT * 2 * 4);
  int*   idx2      = (int*)alloc((size_t)TT * 2 * 4);
  float* bias0     = (float*)alloc((size_t)EE * DD * 4);
  float* bias_sum  = (float*)alloc((size_t)KK * DD * 4);

  auto cdiv = [](int a, int b) { return (a + b - 1) / b; };
  const float* np = nullptr;
  const int*   ni = nullptr;

  // patch embedding
  vit_patchify<<<cdiv(BB * NP * PD, 256), 256, 0, stream>>>(x, patches);
  vit_gemm<GM_BIAS><<<dim3(cdiv(BB * NP, BM), DD / BN), 128, 0, stream>>>(
      patches, patch_w, patch_b, Yp, BB * NP, DD, PD, PD, DD,
      ni, np, ni, np, np, nullptr);
  vit_assemble<<<cdiv(TT * DD, 256), 256, 0, stream>>>(Yp, cls_tok, pos, t);

  for (int l = 0; l < LL; ++l) {
    // attention
    vit_gemm<GM_BIAS><<<dim3(cdiv(TT, BM), QKVD / BN), 128, 0, stream>>>(
        t, qkv_w + (long)l * DD * QKVD, qkv_b + l * QKVD, qkv,
        TT, QKVD, DD, DD, QKVD, ni, np, ni, np, np, nullptr);
    vit_attn_scores<<<BB * NHD * SS, 256, 0, stream>>>(qkv, att);
    vit_av<<<BB * SS, DD, 0, stream>>>(att, qkv, obuf);
    vit_gemm<GM_BIAS><<<dim3(cdiv(TT, BM), DD / BN), 128, 0, stream>>>(
        obuf, proj_w + (long)l * DD * DD, proj_b + l * DD, abuf,
        TT, DD, DD, DD, DD, ni, np, ni, np, np, nullptr);
    vit_add_ln<<<cdiv(TT, 8), 256, 0, stream>>>(t, abuf, ln1_g + l * DD, ln1_b + l * DD, t, TT);

    // MoE: route, then grouped (gathered) expert GEMMs — K/E = 1/4 of dense FLOPs
    hipMemsetAsync(cnt, 0, EE * 4, stream);
    hipMemsetAsync(occ, 0, 2 * EE * 4, stream);
    hipMemsetAsync(moe_out, 0, (size_t)TT * DD * 4, stream);
    vit_router<<<TT, 32, 0, stream>>>(t, router_w + (long)l * DD * EE, router_b + l * EE,
                                      cnt, tok_list, gate_list, occ, gates2, idx2);
    vit_bias0<<<cdiv(EE * DD, 256), 256, 0, stream>>>(
        b1 + (long)l * EE * HID, w2 + (long)l * EE * HID * DD, b2 + (long)l * EE * DD, bias0);
    vit_biassum<<<cdiv(KK * DD, 256), 256, 0, stream>>>(occ, bias0, bias_sum);
    for (int e = 0; e < EE; ++e) {
      const float* w1e = w1 + ((long)l * EE + e) * DD * HID;
      const float* b1e = b1 + ((long)l * EE + e) * HID;
      const float* w2e = w2 + ((long)l * EE + e) * HID * DD;
      const float* b2e = b2 + ((long)l * EE + e) * DD;
      vit_gemm<GM_GATHER_GELU><<<dim3(cdiv(TT, BM), HID / BN), 128, 0, stream>>>(
          t, w1e, b1e, h_buf, TT, HID, DD, DD, HID,
          tok_list + e * TT, np, cnt + e, np, np, nullptr);
      vit_gemm<GM_SCATTER_MOE><<<dim3(cdiv(TT, BM), DD / BN), 128, 0, stream>>>(
          h_buf, w2e, np, nullptr, TT, DD, HID, HID, DD,
          tok_list + e * TT, gate_list + e * TT, cnt + e, bias0 + e * DD, b2e, moe_out);
    }
    vit_moe_biasadd<<<cdiv(TT * DD, 256), 256, 0, stream>>>(gates2, bias_sum, moe_out);
    vit_add_ln<<<cdiv(TT, 8), 256, 0, stream>>>(t, moe_out, ln2_g + l * DD, ln2_b + l * DD, t, TT);
  }

  vit_add_ln<<<cdiv(TT, 8), 256, 0, stream>>>(t, nullptr, norm_g, norm_b, t, TT);
  vit_head<<<cdiv(BB * NCLS, 256), 256, 0, stream>>>(t, fc_w, fc_b, out);
}
